// SingleHeadLongformerModel_18794776887862
// MI455X (gfx1250) — matrix operand order repaired and checked by reading, not compiled, not run on hardware
//
#include <hip/hip_runtime.h>
#include <math.h>
#include <stdint.h>

// ---------------------------------------------------------------------------
// Longformer-style encoder forward for MI455X (gfx1250, wave32, WMMA).
// All GEMM-shaped work runs on v_wmma_f32_16x16x32_f16.
// GEMM staging uses CDNA5 async global->LDS loads (ASYNCcnt) when available.
// ---------------------------------------------------------------------------

typedef __attribute__((ext_vector_type(16))) _Float16 v16h;
typedef __attribute__((ext_vector_type(8)))  _Float16 v8h;
typedef __attribute__((ext_vector_type(8)))  float    v8f;

#define HID  768
#define FFD  3072
#define SEQ  4096
#define NB   2
#define NHD  12
#define HDIM 64
#define WIN  256
#define NLAY 12
#define MTOK (NB * SEQ)   // 8192 rows, divisible by 128

// ---- CDNA5 async global->LDS path (guarded probe) -------------------------
#if defined(__has_builtin)
#if __has_builtin(__builtin_amdgcn_global_load_async_to_lds_b128) && \
    __has_builtin(__builtin_amdgcn_s_wait_asynccnt)
#define USE_ASYNC 1
#endif
#endif
#ifndef USE_ASYNC
#define USE_ASYNC 0
#endif

#if USE_ASYNC
// Builtin signature (from clang diagnostics): parameters are pointers to a
// 16-byte int vector in addrspace(1) (global) and addrspace(3) (LDS).
typedef int v4i_async __attribute__((__vector_size__(16)));
typedef __attribute__((address_space(1))) v4i_async* gas_ptr;
typedef __attribute__((address_space(3))) v4i_async* las_ptr;

static __device__ inline gas_ptr to_global(const void* p) {
  return (gas_ptr)(uintptr_t)p;
}
static __device__ inline las_ptr to_lds(const void* p) {
  // generic pointers to LDS carry the LDS offset in the low 32 bits
  return (las_ptr)(unsigned int)(uintptr_t)p;
}
#endif

static __device__ inline v8f wmma16(v16h a, v16h b, v8f c) {
  return __builtin_amdgcn_wmma_f32_16x16x32_f16(false, a, false, b, (short)0, c,
                                                false, false);
}

static __device__ inline v16h join16(v8h lo, v8h hi) {
  v16h r;
  for (int i = 0; i < 8; ++i) { r[i] = lo[i]; r[8 + i] = hi[i]; }
  return r;
}

// ---------------------------------------------------------------------------
// Weight convert + transpose: Wt[n][k] = (f16) W[k][n]
// ---------------------------------------------------------------------------
__global__ void wconvT_kernel(const float* __restrict__ W, _Float16* __restrict__ Wt,
                              int K, int N) {
  size_t i = (size_t)blockIdx.x * 256 + threadIdx.x;
  if (i >= (size_t)K * N) return;
  int n = (int)(i / K);
  int k = (int)(i % K);
  Wt[i] = (_Float16)W[(size_t)k * N + n];
}

// ---------------------------------------------------------------------------
// WMMA GEMM: C[M,N] = A[M,K] (f16, row major) @ Wt[N,K]^T + bias
// 256 threads = 8 waves; 128x128 macro tile; BK = 32; LDS double buffered.
// Each wave owns a 32x64 sub-tile (2 M-tiles x 4 N-tiles): per K-step
// 6 LDS fragment loads feed 8 WMMAs.
// M % 128 == 0, N % 128 == 0, K % 32 == 0 (true for all shapes here).
// ---------------------------------------------------------------------------
template <bool GELU>
__global__ __launch_bounds__(256)
void gemm_wmma_kernel(const _Float16* __restrict__ A, const _Float16* __restrict__ Wt,
                      const float* __restrict__ bias,
                      float* __restrict__ outF, _Float16* __restrict__ outH,
                      int M, int N, int K) {
  __shared__ _Float16 As[2][128 * 32];
  __shared__ _Float16 Bs[2][128 * 32];

  const int m0 = blockIdx.y * 128;
  const int n0 = blockIdx.x * 128;
  const int t = threadIdx.x;
  const int wave = t >> 5;
  const int lane = t & 31;
  const int lrow = lane & 15;   // M (A) / N (B) index within fragment
  const int lhi  = lane >> 4;   // half-wave select
  const int wr   = wave >> 1;   // 0..3 : which 32-row slice
  const int wc   = wave & 1;    // 0..1 : which 64-col slice

  v8f acc[2][4];
  for (int m = 0; m < 2; ++m)
    for (int n = 0; n < 4; ++n)
      for (int j = 0; j < 8; ++j) acc[m][n][j] = 0.0f;

  // staging map: 4 lanes x 16B per 64B row; a wave covers 8 contiguous rows
  const int srow  = t >> 2;         // 0..63
  const int spart = (t & 3) * 8;    // f16 element offset within row

  auto stage = [&](int buf, int k0) {
#if USE_ASYNC
    __builtin_amdgcn_global_load_async_to_lds_b128(
        to_global(A + (size_t)(m0 + srow) * K + k0 + spart),
        to_lds(&As[buf][srow * 32 + spart]), 0, 0);
    __builtin_amdgcn_global_load_async_to_lds_b128(
        to_global(A + (size_t)(m0 + 64 + srow) * K + k0 + spart),
        to_lds(&As[buf][(64 + srow) * 32 + spart]), 0, 0);
    __builtin_amdgcn_global_load_async_to_lds_b128(
        to_global(Wt + (size_t)(n0 + srow) * K + k0 + spart),
        to_lds(&Bs[buf][srow * 32 + spart]), 0, 0);
    __builtin_amdgcn_global_load_async_to_lds_b128(
        to_global(Wt + (size_t)(n0 + 64 + srow) * K + k0 + spart),
        to_lds(&Bs[buf][(64 + srow) * 32 + spart]), 0, 0);
#else
    *(v8h*)(&As[buf][srow * 32 + spart]) =
        *(const v8h*)(A + (size_t)(m0 + srow) * K + k0 + spart);
    *(v8h*)(&As[buf][(64 + srow) * 32 + spart]) =
        *(const v8h*)(A + (size_t)(m0 + 64 + srow) * K + k0 + spart);
    *(v8h*)(&Bs[buf][srow * 32 + spart]) =
        *(const v8h*)(Wt + (size_t)(n0 + srow) * K + k0 + spart);
    *(v8h*)(&Bs[buf][(64 + srow) * 32 + spart]) =
        *(const v8h*)(Wt + (size_t)(n0 + 64 + srow) * K + k0 + spart);
#endif
  };

  auto compute = [&](int buf) {
    const _Float16* A0 = &As[buf][0];
    const _Float16* B0 = &Bs[buf][0];
    v16h af[2];
#pragma unroll
    for (int mm = 0; mm < 2; ++mm) {
      const _Float16* ap = A0 + (wr * 32 + mm * 16 + lrow) * 32 + lhi * 16;
      af[mm] = join16(*(const v8h*)ap, *(const v8h*)(ap + 8));
    }
#pragma unroll
    for (int nt = 0; nt < 4; ++nt) {
      v16h bf = *(const v16h*)(B0 + (wc * 64 + nt * 16 + lrow) * 32 + lhi * 16);
#pragma unroll
      for (int mm = 0; mm < 2; ++mm)
        acc[mm][nt] = wmma16(af[mm], bf, acc[mm][nt]);
    }
  };

#if USE_ASYNC
  stage(0, 0);
  int cur = 0;
  for (int k0 = 0; k0 < K; k0 += 32) {
    __builtin_amdgcn_s_wait_asynccnt(0);  // own stage of tile `cur` landed
    __syncthreads();                      // all waves staged; prev compute done
    if (k0 + 32 < K) stage(cur ^ 1, k0 + 32);  // overlap next tile with WMMAs
    compute(cur);
    cur ^= 1;
  }
#else
  for (int k0 = 0; k0 < K; k0 += 32) {
    __syncthreads();
    stage(0, k0);
    __syncthreads();
    compute(0);
  }
#endif

  // Epilogue: VGPR r holds row (r + lhi*8), col = lrow within each 16x16 tile
#pragma unroll
  for (int mm = 0; mm < 2; ++mm) {
    const int row0 = m0 + wr * 32 + mm * 16 + lhi * 8;
#pragma unroll
    for (int nt = 0; nt < 4; ++nt) {
      const int col = n0 + wc * 64 + nt * 16 + lrow;
      const float bv = bias ? bias[col] : 0.0f;
#pragma unroll
      for (int r = 0; r < 8; ++r) {
        float x = acc[mm][nt][r] + bv;
        if (GELU) x = 0.5f * x * (1.0f + erff(x * 0.70710678118654752f));
        const size_t off = (size_t)(row0 + r) * N + col;
        if (outF) outF[off] = x;
        if (outH) outH[off] = (_Float16)x;
      }
    }
  }
}

// ---------------------------------------------------------------------------
// Sliding-window attention. One 128-thread block (4 waves) per
// (batch, head, chunk, 16-query tile). Band = 3*WIN = 768 keys.
// ---------------------------------------------------------------------------
#define ATTN_LDS (2048 + 49152 + 24576 + 4096 + 512 + 64)

__global__ __launch_bounds__(128)
void attn_kernel(const _Float16* __restrict__ qb, const _Float16* __restrict__ kb,
                 const _Float16* __restrict__ vb, const int* __restrict__ amask,
                 _Float16* __restrict__ ob) {
  extern __shared__ char smem[];
  _Float16* qs = (_Float16*)smem;                              // 16x64
  float*    sc = (float*)(smem + 2048);                        // 16x768 f32
  _Float16* ps = (_Float16*)(smem + 2048 + 49152);             // 16x768 f16
  _Float16* vt = (_Float16*)(smem + 2048 + 49152 + 24576);     // 64x32 (dim-major)
  float*    red = (float*)(smem + 2048 + 49152 + 24576 + 4096);// 16x8
  float*    rsum = red + 128;                                  // 16

  const int bid  = blockIdx.x;
  const int mt   = bid & 15;          // 16-query tile within chunk
  const int c    = (bid >> 4) & 15;   // chunk
  const int head = (bid >> 8) % NHD;
  const int b    = bid / (NHD * 16 * 16);

  const int t = threadIdx.x;
  const int wave = t >> 5;
  const int lane = t & 31;
  const int lrow = lane & 15;
  const int lhi  = lane >> 4;

  const int qrow0  = c * WIN + mt * 16;   // first seq row of this tile
  const int kstart = c * WIN - WIN;       // band position t=0 -> key index

  // ---- load Q tile (16x64), pre-scaled by 1/sqrt(D) ----
  {
    const int r = t >> 3, d0 = (t & 7) * 8;
    v8h v = *(const v8h*)(qb + ((size_t)(b * SEQ + qrow0 + r)) * HID +
                          head * HDIM + d0);
    for (int i = 0; i < 8; ++i) v[i] = (_Float16)((float)v[i] * 0.125f);
    *(v8h*)(qs + r * 64 + d0) = v;
  }
  __syncthreads();

  // ---- scores: each wave computes 12 of the 48 16-key tiles ----
  for (int kt = wave * 12; kt < wave * 12 + 12; ++kt) {
    v8f acc;
    for (int j = 0; j < 8; ++j) acc[j] = 0.0f;
    const int key = kstart + kt * 16 + lrow;
    const bool kvalid = (key >= 0) && (key < SEQ);
#pragma unroll
    for (int kk = 0; kk < 2; ++kk) {
      const _Float16* ap = qs + lrow * 64 + kk * 32 + lhi * 16;
      v16h af = join16(*(const v8h*)ap, *(const v8h*)(ap + 8));
      v16h bf;
      if (kvalid) {
        bf = *(const v16h*)(kb + ((size_t)(b * SEQ + key)) * HID +
                            head * HDIM + kk * 32 + lhi * 16);
      } else {
        for (int i = 0; i < 16; ++i) bf[i] = (_Float16)0.0f;
      }
      acc = wmma16(af, bf, acc);
    }
    for (int r = 0; r < 8; ++r)
      sc[(r + lhi * 8) * 768 + kt * 16 + lrow] = acc[r];
  }
  __syncthreads();

  // ---- masked softmax over 768-wide band (8 threads per row) ----
  {
    const int r = t >> 3, j = t & 7;
    const int rq = mt * 16 + r;           // query index within chunk
    float mx = -1e30f;
    for (int col = j; col < 768; col += 8) {
      const int kp = kstart + col;
      const bool ok = (col >= rq) && (col <= rq + 2 * WIN) &&
                      (kp >= 0) && (kp < SEQ) && (amask[b * SEQ + kp] > 0);
      const float v = ok ? sc[r * 768 + col] : -1e9f;
      sc[r * 768 + col] = v;
      mx = fmaxf(mx, v);
    }
    red[r * 8 + j] = mx;
    __syncthreads();
    mx = red[r * 8];
    for (int i = 1; i < 8; ++i) mx = fmaxf(mx, red[r * 8 + i]);
    __syncthreads();
    float sum = 0.0f;
    for (int col = j; col < 768; col += 8) {
      const float e = __expf(sc[r * 768 + col] - mx);
      ps[r * 768 + col] = (_Float16)e;   // unnormalized probabilities
      sum += e;
    }
    red[r * 8 + j] = sum;
    __syncthreads();
    if (j == 0) {
      float s = 0.0f;
      for (int i = 0; i < 8; ++i) s += red[r * 8 + i];
      rsum[r] = s;
    }
  }
  __syncthreads();

  // ---- PV: each wave owns one 16-dim output tile, loops 24 key blocks ----
  v8f oacc;
  for (int j = 0; j < 8; ++j) oacc[j] = 0.0f;

  for (int kb2 = 0; kb2 < 24; ++kb2) {
    __syncthreads();
    {   // stage V block transposed: vt[d][kk], 32 keys x 64 dims
      const int key = t >> 2, dpart = (t & 3) * 16;
      const int gk = kstart + kb2 * 32 + key;
      v16h v;
      if (gk >= 0 && gk < SEQ) {
        v = *(const v16h*)(vb + ((size_t)(b * SEQ + gk)) * HID +
                           head * HDIM + dpart);
      } else {
        for (int i = 0; i < 16; ++i) v[i] = (_Float16)0.0f;
      }
      for (int i = 0; i < 16; ++i) vt[(dpart + i) * 32 + key] = v[i];
    }
    __syncthreads();

    const _Float16* ap = ps + lrow * 768 + kb2 * 32 + lhi * 16;
    v16h af = join16(*(const v8h*)ap, *(const v8h*)(ap + 8));
    v16h bf = *(const v16h*)(vt + (wave * 16 + lrow) * 32 + lhi * 16);
    oacc = wmma16(af, bf, oacc);
  }

  {   // normalize by row sum and write f16 output [B,S,NH*D]
    const int sr = qrow0 + lhi * 8;
    for (int r = 0; r < 8; ++r) {
      const float inv = 1.0f / rsum[r + lhi * 8];
      ob[((size_t)(b * SEQ + sr + r)) * HID + head * HDIM + wave * 16 + lrow] =
          (_Float16)(oacc[r] * inv);
    }
  }
}

// ---------------------------------------------------------------------------
// Embedding + LayerNorm (one block per token)
// ---------------------------------------------------------------------------
__global__ __launch_bounds__(256)
void embed_ln_kernel(const int* __restrict__ ids, const float* __restrict__ we,
                     const float* __restrict__ pe, const float* __restrict__ te,
                     const float* __restrict__ g, const float* __restrict__ bt,
                     float* __restrict__ h, _Float16* __restrict__ hb) {
  const int tok = blockIdx.x;
  const int s = tok & (SEQ - 1);
  const int id = ids[tok];
  const int t = threadIdx.x;
  __shared__ float buf[HID];
  __shared__ float red[256];

  float sum = 0.0f;
  for (int i = 0; i < 3; ++i) {
    const int idx = t + i * 256;
    const float v = we[(size_t)id * HID + idx] + pe[(size_t)(s + 2) * HID + idx] + te[idx];
    buf[idx] = v;
    sum += v;
  }
  red[t] = sum;
  __syncthreads();
  for (int o = 128; o > 0; o >>= 1) { if (t < o) red[t] += red[t + o]; __syncthreads(); }
  const float mean = red[0] / (float)HID;
  __syncthreads();
  float sq = 0.0f;
  for (int i = 0; i < 3; ++i) {
    const float d = buf[t + i * 256] - mean;
    sq += d * d;
  }
  red[t] = sq;
  __syncthreads();
  for (int o = 128; o > 0; o >>= 1) { if (t < o) red[t] += red[t + o]; __syncthreads(); }
  const float rstd = rsqrtf(red[0] / (float)HID + 1e-5f);
  for (int i = 0; i < 3; ++i) {
    const int idx = t + i * 256;
    const float y = (buf[idx] - mean) * rstd * g[idx] + bt[idx];
    h[(size_t)tok * HID + idx] = y;
    hb[(size_t)tok * HID + idx] = (_Float16)y;
  }
}

// ---------------------------------------------------------------------------
// residual add + LayerNorm (one block per token)
// ---------------------------------------------------------------------------
__global__ __launch_bounds__(256)
void add_ln_kernel(const float* __restrict__ hin, const float* __restrict__ tmp,
                   const float* __restrict__ g, const float* __restrict__ bt,
                   float* __restrict__ hout, _Float16* __restrict__ hb) {
  const int tok = blockIdx.x;
  const int t = threadIdx.x;
  __shared__ float buf[HID];
  __shared__ float red[256];

  float sum = 0.0f;
  for (int i = 0; i < 3; ++i) {
    const int idx = t + i * 256;
    const float v = hin[(size_t)tok * HID + idx] + tmp[(size_t)tok * HID + idx];
    buf[idx] = v;
    sum += v;
  }
  red[t] = sum;
  __syncthreads();
  for (int o = 128; o > 0; o >>= 1) { if (t < o) red[t] += red[t + o]; __syncthreads(); }
  const float mean = red[0] / (float)HID;
  __syncthreads();
  float sq = 0.0f;
  for (int i = 0; i < 3; ++i) {
    const float d = buf[t + i * 256] - mean;
    sq += d * d;
  }
  red[t] = sq;
  __syncthreads();
  for (int o = 128; o > 0; o >>= 1) { if (t < o) red[t] += red[t + o]; __syncthreads(); }
  const float rstd = rsqrtf(red[0] / (float)HID + 1e-5f);
  for (int i = 0; i < 3; ++i) {
    const int idx = t + i * 256;
    const float y = (buf[idx] - mean) * rstd * g[idx] + bt[idx];
    hout[(size_t)tok * HID + idx] = y;
    hb[(size_t)tok * HID + idx] = (_Float16)y;
  }
}

// ---------------------------------------------------------------------------
// Classifier head: out[b,c] = relu(h[b,0,:]@w1+b1) @ w2 + b2   (tiny)
// ---------------------------------------------------------------------------
__global__ __launch_bounds__(512)
void cls_kernel(const float* __restrict__ h, const float* __restrict__ w1,
                const float* __restrict__ b1, const float* __restrict__ w2,
                const float* __restrict__ b2, float* __restrict__ out) {
  __shared__ float x[2 * 512];
  const int t = threadIdx.x;
  for (int b = 0; b < 2; ++b) {
    const float* p = h + (size_t)b * SEQ * HID;   // h[b,0,:]
    float s = b1[t];
    for (int k = 0; k < HID; ++k) s += p[k] * w1[(size_t)k * 512 + t];
    x[b * 512 + t] = fmaxf(s, 0.0f);
  }
  __syncthreads();
  if (t < 2 * 43) {
    const int b = t / 43, cc = t % 43;
    float s = b2[cc];
    for (int k = 0; k < 512; ++k) s += x[b * 512 + k] * w2[(size_t)k * 43 + cc];
    out[b * 43 + cc] = s;
  }
}

// ---------------------------------------------------------------------------
// Host orchestration
// ---------------------------------------------------------------------------
extern "C" void kernel_launch(void* const* d_in, const int* in_sizes, int n_in,
                              void* d_out, int out_size, void* d_ws, size_t ws_size,
                              hipStream_t stream) {
  const int*   ids   = (const int*)d_in[0];
  const int*   amask = (const int*)d_in[1];
  const float* we    = (const float*)d_in[2];
  const float* pe    = (const float*)d_in[3];
  const float* te    = (const float*)d_in[4];
  const float* eg    = (const float*)d_in[5];
  const float* eb    = (const float*)d_in[6];
  const float* Wq    = (const float*)d_in[7];
  const float* bq    = (const float*)d_in[8];
  const float* Wk    = (const float*)d_in[9];
  const float* bk    = (const float*)d_in[10];
  const float* Wv    = (const float*)d_in[11];
  const float* bv    = (const float*)d_in[12];
  const float* Wo    = (const float*)d_in[13];
  const float* bo    = (const float*)d_in[14];
  const float* g1    = (const float*)d_in[15];
  const float* b1    = (const float*)d_in[16];
  const float* Wi    = (const float*)d_in[17];
  const float* bi    = (const float*)d_in[18];
  const float* Wo2   = (const float*)d_in[19];
  const float* bo2   = (const float*)d_in[20];
  const float* g2    = (const float*)d_in[21];
  const float* b2    = (const float*)d_in[22];
  const float* cw1   = (const float*)d_in[23];
  const float* cb1   = (const float*)d_in[24];
  const float* cw2   = (const float*)d_in[25];
  const float* cb2   = (const float*)d_in[26];

  // workspace carve-up
  char* ws = (char*)d_ws;
  size_t off = 0;
  auto carve = [&](size_t bytes) -> void* {
    void* p = ws + off;
    off = (off + bytes + 255) & ~(size_t)255;
    return p;
  };
  float*     h    = (float*)    carve((size_t)MTOK * HID * 4);
  float*     tmp  = (float*)    carve((size_t)MTOK * HID * 4);
  _Float16*  hb   = (_Float16*) carve((size_t)MTOK * HID * 2);
  _Float16*  qbuf = (_Float16*) carve((size_t)MTOK * HID * 2);
  _Float16*  kbuf = (_Float16*) carve((size_t)MTOK * HID * 2);
  _Float16*  vbuf = (_Float16*) carve((size_t)MTOK * HID * 2);
  _Float16*  abuf = (_Float16*) carve((size_t)MTOK * HID * 2);
  _Float16*  fbuf = (_Float16*) carve((size_t)MTOK * FFD * 2);
  _Float16*  wt   = (_Float16*) carve((size_t)HID * FFD * 2);
  (void)ws_size; (void)in_sizes; (void)n_in; (void)out_size;

  const dim3 gHH((HID * HID + 255) / 256);
  const dim3 gHF((HID * FFD + 255) / 256);
  const dim3 gemmH(HID / 128, MTOK / 128);   // (6, 64)
  const dim3 gemmF(FFD / 128, MTOK / 128);   // (24, 64)
  const int  attnBlocks = NB * NHD * (SEQ / WIN) * (WIN / 16);  // 6144

  embed_ln_kernel<<<MTOK, 256, 0, stream>>>(ids, we, pe, te, eg, eb, h, hb);

  for (int l = 0; l < NLAY; ++l) {
    const size_t oHH = (size_t)l * HID * HID;
    const size_t oHF = (size_t)l * HID * FFD;

    // Q, K, V projections
    wconvT_kernel<<<gHH, 256, 0, stream>>>(Wq + oHH, wt, HID, HID);
    gemm_wmma_kernel<false><<<gemmH, 256, 0, stream>>>(hb, wt, bq + (size_t)l * HID,
                                                       nullptr, qbuf, MTOK, HID, HID);
    wconvT_kernel<<<gHH, 256, 0, stream>>>(Wk + oHH, wt, HID, HID);
    gemm_wmma_kernel<false><<<gemmH, 256, 0, stream>>>(hb, wt, bk + (size_t)l * HID,
                                                       nullptr, kbuf, MTOK, HID, HID);
    wconvT_kernel<<<gHH, 256, 0, stream>>>(Wv + oHH, wt, HID, HID);
    gemm_wmma_kernel<false><<<gemmH, 256, 0, stream>>>(hb, wt, bv + (size_t)l * HID,
                                                       nullptr, vbuf, MTOK, HID, HID);

    // banded attention
    attn_kernel<<<attnBlocks, 128, ATTN_LDS, stream>>>(qbuf, kbuf, vbuf, amask, abuf);

    // output projection + residual LN
    wconvT_kernel<<<gHH, 256, 0, stream>>>(Wo + oHH, wt, HID, HID);
    gemm_wmma_kernel<false><<<gemmH, 256, 0, stream>>>(abuf, wt, bo + (size_t)l * HID,
                                                       tmp, nullptr, MTOK, HID, HID);
    add_ln_kernel<<<MTOK, 256, 0, stream>>>(h, tmp, g1 + (size_t)l * HID,
                                            b1 + (size_t)l * HID, h, hb);

    // FFN: GELU fused into first GEMM epilogue
    wconvT_kernel<<<gHF, 256, 0, stream>>>(Wi + oHF, wt, HID, FFD);
    gemm_wmma_kernel<true><<<gemmF, 256, 0, stream>>>(hb, wt, bi + (size_t)l * FFD,
                                                      nullptr, fbuf, MTOK, FFD, HID);
    wconvT_kernel<<<gHF, 256, 0, stream>>>(Wo2 + oHF, wt, FFD, HID);
    gemm_wmma_kernel<false><<<gemmH, 256, 0, stream>>>(fbuf, wt, bo2 + (size_t)l * HID,
                                                       tmp, nullptr, MTOK, HID, FFD);
    add_ln_kernel<<<MTOK, 256, 0, stream>>>(h, tmp, g2 + (size_t)l * HID,
                                            b2 + (size_t)l * HID, h, hb);
  }

  cls_kernel<<<1, 512, 0, stream>>>(h, cw1, cb1, cw2, cb2, (float*)d_out);
}
